// SFTLayer_86715389706726
// MI455X (gfx1250) — matrix operand order repaired and checked by reading, not compiled, not run on hardware
//
#include <hip/hip_runtime.h>

typedef __attribute__((ext_vector_type(16))) _Float16 v16h;
typedef __attribute__((ext_vector_type(8)))  _Float16 v8h;
typedef __attribute__((ext_vector_type(2)))  _Float16 v2h;
typedef __attribute__((ext_vector_type(8)))  float    v8f;
typedef __attribute__((ext_vector_type(4)))  float    v4f;

#define NB     2
#define NH     128
#define NW     128
#define NBANDS 32
#define NC     64      // Cout == CPSI == 64
#define NTAPS  9
#define LEAKY  0.3f

// ---------------------------------------------------------------------------
// Prep: convert conv weights [tap(9)][cin(64)][cout(64)] f32 (HWIO) into
// f16, transposed to [field(2)][tap(9)][cout(64)][cin(64)] so that WMMA
// B-fragments are contiguous along cin (K). 147456 B in d_ws.
// ---------------------------------------------------------------------------
__global__ __launch_bounds__(256)
void sft_prep_weights(const float* __restrict__ gw,
                      const float* __restrict__ bw,
                      _Float16* __restrict__ wt) {
  int id = blockIdx.x * 256 + threadIdx.x;
  if (id >= 2 * NTAPS * NC * NC) return;
  int f   = id / (NTAPS * NC * NC);
  int r   = id % (NTAPS * NC * NC);
  int tap = r / (NC * NC);
  int r2  = r % (NC * NC);
  int co  = r2 / NC;
  int ci  = r2 % NC;
  const float* src = f ? bw : gw;                 // [tap][cin][cout]
  wt[id] = (_Float16)src[(tap * NC + ci) * NC + co];
}

__device__ __forceinline__ v16h cat16(v8h lo, v8h hi) {
  return __builtin_shufflevector(lo, hi, 0,1,2,3,4,5,6,7,8,9,10,11,12,13,14,15);
}

// ---------------------------------------------------------------------------
// Fused kernel: per workgroup (256 threads = 8 waves), 32 pixels of one row.
//   phase 1: stage psi patch [3][34][64] -> LDS f16 (zero-padded SAME)
//   phase 2: implicit-GEMM conv via v_wmma_f32_16x16x32_f16
//            wave = field(2) x n-tile(4); M=2x16 pixels, K=9*64 in 18 chunks
//   phase 3: FiLM broadcast, NT-streamed float4, gamma/beta from LDS
// ---------------------------------------------------------------------------
__global__ __launch_bounds__(256)
void sft_main(const float* __restrict__ x,
              const float* __restrict__ psi,
              const float* __restrict__ gbias,
              const float* __restrict__ bbias,
              const _Float16* __restrict__ wt,
              float* __restrict__ out) {
  __shared__ __align__(16) _Float16 lp[3 * 34 * NC];   // 13056 B psi patch
  __shared__ __align__(16) float    gb[2 * 32 * NC];   // 16384 B gamma/beta

  const int tid  = threadIdx.x;
  const int tile = blockIdx.x;                 // 0..1023
  const int b    = tile >> 9;                  // / (128 rows * 4 tiles)
  const int h    = (tile >> 2) & (NH - 1);
  const int w0   = (tile & 3) << 5;            // 32-pixel W tile

  // ---- phase 1: stage psi patch (f32 -> f16), SAME zero padding ----
  for (int e = tid; e < 3 * 34 * (NC / 2); e += 256) {
    int ty   = e / (34 * 32);
    int r    = e - ty * (34 * 32);
    int colp = r >> 5;
    int cp   = (r & 31) << 1;
    int hh   = h - 1 + ty;
    int ww   = w0 - 1 + colp;
    _Float16 a0 = (_Float16)0.f, a1 = (_Float16)0.f;
    if (hh >= 0 && hh < NH && ww >= 0 && ww < NW) {
      const float* s = psi + (((size_t)(b * NH + hh) * NW + ww) * NC + cp);
      a0 = (_Float16)s[0];
      a1 = (_Float16)s[1];
    }
    v2h v; v.x = a0; v.y = a1;
    *(v2h*)&lp[(ty * 34 + colp) * NC + cp] = v;
  }
  __syncthreads();

  // ---- phase 2: WMMA conv ----
  const int lane  = tid & 31;
  const int wv    = tid >> 5;
  const int field = wv >> 2;         // 0 = gamma, 1 = beta
  const int nt    = wv & 3;          // 16-wide cout tile
  const int half  = lane >> 4;
  const int lrow  = lane & 15;       // M row (A) / N col (B,C,D)
  const int cout  = nt * 16 + lrow;

  v8f acc0 = {0.f,0.f,0.f,0.f,0.f,0.f,0.f,0.f};
  v8f acc1 = acc0;
  const _Float16* wfld = wt + (size_t)field * (NTAPS * NC * NC);

  for (int c = 0; c < 18; ++c) {               // K chunks of 32
    const int tap  = c >> 1;
    const int ty   = tap / 3;
    const int tx   = tap - ty * 3;
    const int koff = (c & 1) * 32;             // cin half of this tap
    // B 32x16 fragment: half 0: K={0..7,16..23}; half 1: K={8..15,24..31}
    const _Float16* bp = wfld + ((size_t)(tap * NC + cout) * NC + koff + half * 8);
    v16h bf = cat16(*(const v8h*)bp, *(const v8h*)(bp + 16));
    // A 16x32 fragment: half 0: K={0..7,16..23}; half 1: K={8..15,24..31}
    const int acin = koff + half * 8;
    {
      const _Float16* ap = &lp[(ty * 34 + (lrow + tx)) * NC + acin];
      v16h af = cat16(*(const v8h*)ap, *(const v8h*)(ap + 16));
      acc0 = __builtin_amdgcn_wmma_f32_16x16x32_f16(false, af, false, bf,
                                                    (short)0, acc0, false, false);
    }
    {
      const _Float16* ap = &lp[(ty * 34 + (16 + lrow + tx)) * NC + acin];
      v16h af = cat16(*(const v8h*)ap, *(const v8h*)(ap + 16));
      acc1 = __builtin_amdgcn_wmma_f32_16x16x32_f16(false, af, false, bf,
                                                    (short)0, acc1, false, false);
    }
  }

  // bias + LeakyReLU, park results in LDS. D layout: VGPR r -> M = r + 8*half
  const float bias = (field ? bbias : gbias)[cout];
  #pragma unroll
  for (int r = 0; r < 8; ++r) {
    const int m = r + half * 8;
    float v0 = acc0[r] + bias;
    v0 = v0 > 0.f ? v0 : LEAKY * v0;
    gb[field * (32 * NC) + m * NC + cout] = v0;
    float v1 = acc1[r] + bias;
    v1 = v1 > 0.f ? v1 : LEAKY * v1;
    gb[field * (32 * NC) + (m + 16) * NC + cout] = v1;
  }
  __syncthreads();

  // ---- phase 3: FiLM broadcast (memory-bound; NT streaming) ----
  const size_t base = ((size_t)(b * NH + h) * NW + w0) * (NBANDS * NC);
  const v4f* xv = (const v4f*)(x + base);
  v4f*       ov = (v4f*)(out + base);
  for (int v = tid; v < 32 * NBANDS * NC / 4; v += 256) {
    const int flat = v << 2;
    const int i    = flat >> 11;               // pixel in tile
    const int cc   = flat & (NC - 1);          // channel quad
    v4f g  = *(const v4f*)&gb[i * NC + cc];
    v4f be = *(const v4f*)&gb[32 * NC + i * NC + cc];
    v4f xx = __builtin_nontemporal_load(&xv[v]);
    __builtin_nontemporal_store(g * xx + be, &ov[v]);
  }
}

extern "C" void kernel_launch(void* const* d_in, const int* in_sizes, int n_in,
                              void* d_out, int out_size, void* d_ws, size_t ws_size,
                              hipStream_t stream) {
  (void)in_sizes; (void)n_in; (void)out_size; (void)ws_size;
  const float* x   = (const float*)d_in[0];
  const float* psi = (const float*)d_in[1];
  const float* gw  = (const float*)d_in[2];
  const float* gbb = (const float*)d_in[3];
  const float* bw  = (const float*)d_in[4];
  const float* bbb = (const float*)d_in[5];
  _Float16* wt = (_Float16*)d_ws;              // 2*9*64*64 f16 = 147456 B

  sft_prep_weights<<<(2 * NTAPS * NC * NC) / 256, 256, 0, stream>>>(gw, bw, wt);
  sft_main<<<NB * NH * (NW / 32), 256, 0, stream>>>(x, psi, gbb, bbb, wt,
                                                    (float*)d_out);
}
